// EncoderLayer_87024627352370
// MI455X (gfx1250) — matrix operand order repaired and checked by reading, not compiled, not run on hardware
//
#include <hip/hip_runtime.h>
#include <stdint.h>

// ---------------------------------------------------------------------------
// Transformer encoder layer for MI455X (gfx1250, wave32, WMMA).
// B=2, S=2048, D=1024, H=16, DH=64, FF=4096.  All GEMMs run on
// v_wmma_f32_16x16x32_bf16; softmax / layernorm epilogues in fp32.
// ---------------------------------------------------------------------------

#define B_  2
#define S_  2048
#define D_  1024
#define H_  16
#define DH_ 64
#define FF_ 4096
#define EPS_ 1e-6f

typedef __bf16 bf16;
typedef __attribute__((ext_vector_type(16))) __bf16 v16bf;
typedef __attribute__((ext_vector_type(8)))  __bf16 v8bf;
typedef __attribute__((ext_vector_type(8)))  float  v8f;

__device__ __forceinline__ v8f wmma_bf16(v16bf a, v16bf b, v8f c) {
  // (neg_a, A, neg_b, B, c_mod, C, reuse_a, reuse_b)
  return __builtin_amdgcn_wmma_f32_16x16x32_bf16(false, a, false, b,
                                                 (short)0, c, false, false);
}

// A fragment (16x32, M x K): lane m = l%16, half h = l/16 holds
// k in [8h,8h+7] and [16+8h, 23+8h]  (ISA 7.12.2 16-bit A layout).
// Source: row-major [M][K] bf16, 16B-aligned rows.
__device__ __forceinline__ v16bf load_frag_a(const bf16* __restrict__ src,
                                             int ld, int row0, int k0) {
  int lane = threadIdx.x & 31;
  const bf16* p = src + (size_t)(row0 + (lane & 15)) * ld + k0 + 8 * (lane >> 4);
  v8bf lo = *(const v8bf*)p;        // k = 8h .. 8h+7
  v8bf hi = *(const v8bf*)(p + 16); // k = 16+8h .. 23+8h
  v16bf f;
#pragma unroll
  for (int i = 0; i < 8; ++i) { f[i] = lo[i]; f[i + 8] = hi[i]; }
  return f;
}

// B fragment (32x16, K x N): lane n = l%16, half h holds
// k in [8h,8h+7] and [16+8h, 23+8h]  (same k order as A fragment).
// Source: row-major [N][K] bf16 (i.e. transposed weight / K rows / Vt rows).
__device__ __forceinline__ v16bf load_frag_b(const bf16* __restrict__ src,
                                             int ld, int col0, int k0) {
  int lane = threadIdx.x & 31;
  const bf16* p = src + (size_t)(col0 + (lane & 15)) * ld + k0 + 8 * (lane >> 4);
  v8bf lo = *(const v8bf*)p;        // k = 8h .. 8h+7
  v8bf hi = *(const v8bf*)(p + 16); // k = 16+8h .. 23+8h
  v16bf f;
#pragma unroll
  for (int i = 0; i < 8; ++i) { f[i] = lo[i]; f[i + 8] = hi[i]; }
  return f;
}

// ---------------------------------------------------------------------------
// GEMM: C[M,N] = A[M,K](bf16,row-major) * Bt[N,K](bf16,row-major)^T + bias
// Block tile 128x256: 8 waves as 2x4, per-wave tile 64x64 ->
// 16 WMMA per k-step on 8 fragment loads (A,B each reused 4x).
// Fusion variants are compile-time template parameters (no runtime branches).
// ---------------------------------------------------------------------------
template <bool RESID, bool RELU, bool STOREF, bool STOREH>
__global__ __launch_bounds__(256) void gemm_bf16_kernel(
    const bf16* __restrict__ A, const bf16* __restrict__ Bt,
    const float* __restrict__ bias, const float* __restrict__ resid,
    float* __restrict__ Cf, bf16* __restrict__ Ch,
    int M, int N, int K) {
  int lane = threadIdx.x & 31;
  int wid  = threadIdx.x >> 5;
  int wm = wid >> 2;                 // 0..1
  int wn = wid & 3;                  // 0..3
  int row0 = blockIdx.y * 128 + wm * 64;
  int col0 = blockIdx.x * 256 + wn * 64;

  v8f acc[4][4] = {};
  for (int k0 = 0; k0 < K; k0 += 32) {
    v16bf a[4], b[4];
#pragma unroll
    for (int mt = 0; mt < 4; ++mt) a[mt] = load_frag_a(A, K, row0 + 16 * mt, k0);
#pragma unroll
    for (int nt = 0; nt < 4; ++nt) b[nt] = load_frag_b(Bt, K, col0 + 16 * nt, k0);
#pragma unroll
    for (int mt = 0; mt < 4; ++mt)
#pragma unroll
      for (int nt = 0; nt < 4; ++nt)
        acc[mt][nt] = wmma_bf16(a[mt], b[nt], acc[mt][nt]);
  }

  int n = lane & 15, half = lane >> 4;
#pragma unroll
  for (int mt = 0; mt < 4; ++mt)
#pragma unroll
    for (int nt = 0; nt < 4; ++nt) {
      int gn = col0 + 16 * nt + n;
      float bv = bias[gn];
#pragma unroll
      for (int r = 0; r < 8; ++r) {
        int gm = row0 + 16 * mt + 8 * half + r;   // C layout: M = r + 8*half
        float v = acc[mt][nt][r] + bv;
        if (RESID) v += resid[(size_t)gm * N + gn];
        if (RELU)  v = fmaxf(v, 0.0f);
        if (STOREF) Cf[(size_t)gm * N + gn] = v;
        if (STOREH) Ch[(size_t)gm * N + gn] = (bf16)v;
      }
    }
}

// ---------------------------------------------------------------------------
// Flash attention: per wave, 16 queries x full key axis with online softmax.
// scores = (Q K^T)/8 + mask*-1e4 ; out = softmax(scores) V.
// Q,K bf16 [b*S][H*DH]; Vt bf16 [b*H][DH][S]; AO bf16 [b*S][H*DH].
// ---------------------------------------------------------------------------
__global__ __launch_bounds__(256) void attention_kernel(
    const bf16* __restrict__ Q, const bf16* __restrict__ K,
    const bf16* __restrict__ Vt, const int* __restrict__ mask,
    bf16* __restrict__ AO) {
  __shared__ __align__(16) bf16 pbuf[8][16 * 32];  // 1 KB / wave

  int lane = threadIdx.x & 31;
  int wid  = threadIdx.x >> 5;
  int bh = blockIdx.y;
  int b = bh >> 4;                   // H_ == 16
  int h = bh & 15;
  int q0 = blockIdx.x * 128 + wid * 16;

  const bf16* Qb = Q + (size_t)b * S_ * D_ + h * DH_;
  const bf16* Kb = K + (size_t)b * S_ * D_ + h * DH_;
  const bf16* Vb = Vt + (size_t)bh * DH_ * S_;
  const int*  mb = mask + b * S_;

  v16bf qa0 = load_frag_a(Qb, D_, q0, 0);
  v16bf qa1 = load_frag_a(Qb, D_, q0, 32);

  int n = lane & 15, half = lane >> 4;
  float mrow[8], lrow[8];
  v8f acc[4] = {};
#pragma unroll
  for (int r = 0; r < 8; ++r) { mrow[r] = -3.0e38f; lrow[r] = 0.0f; }

  bf16* pw = pbuf[wid];

  for (int key0 = 0; key0 < S_; key0 += 32) {
    // ---- scores: Q(16x64) x K^T(64x32) -> two 16x16 tiles --------------
    v8f sc[2];
#pragma unroll
    for (int nt = 0; nt < 2; ++nt) {
      v16bf kb0 = load_frag_b(Kb, D_, key0 + 16 * nt, 0);
      v16bf kb1 = load_frag_b(Kb, D_, key0 + 16 * nt, 32);
      v8f c = {};
      c = wmma_bf16(qa0, kb0, c);
      c = wmma_bf16(qa1, kb1, c);
      sc[nt] = c;
    }
    float madd0 = (float)mb[key0 + n]      * -10000.0f;
    float madd1 = (float)mb[key0 + 16 + n] * -10000.0f;

    // ---- online softmax over this 32-key slab --------------------------
    float alpha[8];
#pragma unroll
    for (int r = 0; r < 8; ++r) {
      float s0 = sc[0][r] * 0.125f + madd0;
      float s1 = sc[1][r] * 0.125f + madd1;
      float mx = fmaxf(s0, s1);
#pragma unroll
      for (int o = 8; o >= 1; o >>= 1) mx = fmaxf(mx, __shfl_xor(mx, o, 32));
      float mnew = fmaxf(mrow[r], mx);
      alpha[r] = __expf(mrow[r] - mnew);
      mrow[r] = mnew;
      float p0 = __expf(s0 - mnew);
      float p1 = __expf(s1 - mnew);
      sc[0][r] = p0; sc[1][r] = p1;
      float rs = p0 + p1;
#pragma unroll
      for (int o = 8; o >= 1; o >>= 1) rs += __shfl_xor(rs, o, 32);
      lrow[r] = lrow[r] * alpha[r] + rs;
    }

    // ---- P (C-layout) -> LDS row-major 16x32 -> reload as A fragment ---
#pragma unroll
    for (int nt = 0; nt < 2; ++nt)
#pragma unroll
      for (int r = 0; r < 8; ++r)
        pw[(r + 8 * half) * 32 + 16 * nt + n] = (bf16)sc[nt][r];
    asm volatile("s_wait_dscnt 0x0" ::: "memory");   // same-wave LDS RAW
    v16bf pa = load_frag_a(pw, 32, 0, 0);

    // ---- acc = acc*alpha + P(16x32) x V(32x64) -------------------------
#pragma unroll
    for (int t = 0; t < 4; ++t) {
      v16bf vb = load_frag_b(Vb, S_, t * 16, key0);
#pragma unroll
      for (int r = 0; r < 8; ++r) acc[t][r] *= alpha[r];
      acc[t] = wmma_bf16(pa, vb, acc[t]);
    }
  }

  bf16* AOb = AO + (size_t)b * S_ * D_ + h * DH_;
#pragma unroll
  for (int t = 0; t < 4; ++t)
#pragma unroll
    for (int r = 0; r < 8; ++r) {
      float o = acc[t][r] / lrow[r];
      AOb[(size_t)(q0 + r + 8 * half) * D_ + t * 16 + n] = (bf16)o;
    }
}

// ---------------------------------------------------------------------------
// LayerNorm over last dim (D_=1024), one block (256 thr) per row.
// Works in-place (caches its 4 elements in registers). Optional bf16 copy.
// ---------------------------------------------------------------------------
__global__ __launch_bounds__(256) void layernorm_kernel(
    const float* __restrict__ X, const float* __restrict__ gamma,
    const float* __restrict__ beta, float* __restrict__ outF,
    bf16* __restrict__ outH) {
  __shared__ float red[256];
  int row = blockIdx.x;
  int tid = threadIdx.x;
  const float* xr = X + (size_t)row * D_;
  float v[4];
  float s = 0.0f, ss = 0.0f;
#pragma unroll
  for (int i = 0; i < 4; ++i) {
    v[i] = xr[tid + 256 * i];
    s += v[i]; ss += v[i] * v[i];
  }
  red[tid] = s; __syncthreads();
  for (int o = 128; o > 0; o >>= 1) { if (tid < o) red[tid] += red[tid + o]; __syncthreads(); }
  float mu = red[0] * (1.0f / D_);
  __syncthreads();
  red[tid] = ss; __syncthreads();
  for (int o = 128; o > 0; o >>= 1) { if (tid < o) red[tid] += red[tid + o]; __syncthreads(); }
  float var  = red[0] * (1.0f / D_) - mu * mu;
  float rstd = rsqrtf(var + EPS_);
#pragma unroll
  for (int i = 0; i < 4; ++i) {
    int c = tid + 256 * i;
    float y = (v[i] - mu) * rstd * gamma[c] + beta[c];
    if (outF) outF[(size_t)row * D_ + c] = y;
    if (outH) outH[(size_t)row * D_ + c] = (bf16)y;
  }
}

// fp32 -> bf16 elementwise
__global__ __launch_bounds__(256) void f32_to_bf16_kernel(
    const float* __restrict__ s, bf16* __restrict__ d, int n) {
  int i = blockIdx.x * 256 + threadIdx.x;
  if (i < n) d[i] = (bf16)s[i];
}

// fp32 [K][N] -> bf16 [N][K]  (tiled transpose)
__global__ void transpose_f32_bf16_kernel(const float* __restrict__ src,
                                          bf16* __restrict__ dst, int K, int N) {
  __shared__ float t[32][33];
  int k0 = blockIdx.y * 32, n0 = blockIdx.x * 32;
  int tx = threadIdx.x, ty = threadIdx.y;
  for (int i = ty; i < 32; i += 8)
    t[i][tx] = src[(size_t)(k0 + i) * N + n0 + tx];
  __syncthreads();
  for (int i = ty; i < 32; i += 8)
    dst[(size_t)(n0 + i) * K + k0 + tx] = (bf16)t[tx][i];
}

// bf16 V [b*S][H*DH] -> Vt [b*H][DH][S]   (per-head tiled transpose)
__global__ void build_vt_kernel(const bf16* __restrict__ V, bf16* __restrict__ Vt) {
  __shared__ bf16 t[32][33];
  int bh = blockIdx.z;
  int b = bh >> 4, h = bh & 15;
  int s0 = blockIdx.x * 32, d0 = blockIdx.y * 32;
  int tx = threadIdx.x, ty = threadIdx.y;
  const bf16* src = V + (size_t)b * S_ * D_ + h * DH_;
  for (int i = ty; i < 32; i += 8)
    t[i][tx] = src[(size_t)(s0 + i) * D_ + d0 + tx];
  __syncthreads();
  bf16* dst = Vt + (size_t)bh * DH_ * S_;
  for (int i = ty; i < 32; i += 8)
    dst[(size_t)(d0 + i) * S_ + s0 + tx] = t[tx][i];
}

// ---------------------------------------------------------------------------
extern "C" void kernel_launch(void* const* d_in, const int* in_sizes, int n_in,
                              void* d_out, int out_size, void* d_ws, size_t ws_size,
                              hipStream_t stream) {
  (void)in_sizes; (void)n_in; (void)out_size; (void)ws_size;

  const float* X    = (const float*)d_in[0];
  const int*   mask = (const int*)  d_in[1];
  const float* Wq   = (const float*)d_in[2];
  const float* bq   = (const float*)d_in[3];
  const float* Wk   = (const float*)d_in[4];
  const float* bk   = (const float*)d_in[5];
  const float* Wv   = (const float*)d_in[6];
  const float* bv   = (const float*)d_in[7];
  const float* Wo   = (const float*)d_in[8];
  const float* bo   = (const float*)d_in[9];
  const float* g0   = (const float*)d_in[10];
  const float* be0  = (const float*)d_in[11];
  const float* W1   = (const float*)d_in[12];
  const float* b1   = (const float*)d_in[13];
  const float* W2   = (const float*)d_in[14];
  const float* b2   = (const float*)d_in[15];
  const float* g1   = (const float*)d_in[16];
  const float* be1  = (const float*)d_in[17];
  float* out = (float*)d_out;

  char* ws = (char*)d_ws;
  const size_t MB = 1ull << 20;
  bf16*  Xh  = (bf16*) (ws +  0 * MB);   // 8 MB  X bf16
  bf16*  Qh  = (bf16*) (ws +  8 * MB);   // 8 MB
  bf16*  Kh  = (bf16*) (ws + 16 * MB);   // 8 MB
  bf16*  Vh  = (bf16*) (ws + 24 * MB);   // 8 MB
  bf16*  Vt  = (bf16*) (ws + 32 * MB);   // 8 MB  [b*H][DH][S]
  bf16*  AO  = (bf16*) (ws + 40 * MB);   // 8 MB  attention output
  bf16*  WqT = (bf16*) (ws + 48 * MB);   // 2 MB each
  bf16*  WkT = (bf16*) (ws + 50 * MB);
  bf16*  WvT = (bf16*) (ws + 52 * MB);
  bf16*  WoT = (bf16*) (ws + 54 * MB);
  bf16*  W1T = (bf16*) (ws + 56 * MB);   // 8 MB  [FF][D]
  bf16*  W2T = (bf16*) (ws + 64 * MB);   // 8 MB  [D][FF]
  float* X1f = (float*)(ws + 72 * MB);   // 16 MB
  bf16*  X1h = (bf16*) (ws + 88 * MB);   // 8 MB
  bf16*  H1  = (bf16*) (ws + 96 * MB);   // 32 MB  [M][FF]
  // total workspace: 128 MB

  const int M = B_ * S_;                 // 4096 rows
  dim3 blk256(256);
  dim3 tb(32, 8);

  // --- precision / layout prep ---
  f32_to_bf16_kernel<<<dim3((M * D_) / 256), blk256, 0, stream>>>(X, Xh, M * D_);
  transpose_f32_bf16_kernel<<<dim3(D_ / 32,  D_ / 32), tb, 0, stream>>>(Wq, WqT, D_,  D_);
  transpose_f32_bf16_kernel<<<dim3(D_ / 32,  D_ / 32), tb, 0, stream>>>(Wk, WkT, D_,  D_);
  transpose_f32_bf16_kernel<<<dim3(D_ / 32,  D_ / 32), tb, 0, stream>>>(Wv, WvT, D_,  D_);
  transpose_f32_bf16_kernel<<<dim3(D_ / 32,  D_ / 32), tb, 0, stream>>>(Wo, WoT, D_,  D_);
  transpose_f32_bf16_kernel<<<dim3(FF_ / 32, D_ / 32), tb, 0, stream>>>(W1, W1T, D_,  FF_);
  transpose_f32_bf16_kernel<<<dim3(D_ / 32, FF_ / 32), tb, 0, stream>>>(W2, W2T, FF_, D_);

  // --- QKV projections (bf16 outputs) ---
  dim3 gD(D_ / 256, M / 128);            // (4, 32)
  gemm_bf16_kernel<false, false, false, true><<<gD, blk256, 0, stream>>>(
      Xh, WqT, bq, nullptr, nullptr, Qh, M, D_, D_);
  gemm_bf16_kernel<false, false, false, true><<<gD, blk256, 0, stream>>>(
      Xh, WkT, bk, nullptr, nullptr, Kh, M, D_, D_);
  gemm_bf16_kernel<false, false, false, true><<<gD, blk256, 0, stream>>>(
      Xh, WvT, bv, nullptr, nullptr, Vh, M, D_, D_);

  build_vt_kernel<<<dim3(S_ / 32, DH_ / 32, B_ * H_), tb, 0, stream>>>(Vh, Vt);

  // --- attention ---
  attention_kernel<<<dim3(S_ / 128, B_ * H_), blk256, 0, stream>>>(Qh, Kh, Vt, mask, AO);

  // --- out-proj + residual(X) -> d_out (fp32 scratch), then LN0 ---
  gemm_bf16_kernel<true, false, true, false><<<gD, blk256, 0, stream>>>(
      AO, WoT, bo, X, out, nullptr, M, D_, D_);
  layernorm_kernel<<<dim3(M), blk256, 0, stream>>>(out, g0, be0, X1f, X1h);

  // --- FFN ---
  gemm_bf16_kernel<false, true, false, true><<<dim3(FF_ / 256, M / 128), blk256, 0, stream>>>(
      X1h, W1T, b1, nullptr, nullptr, H1, M, FF_, D_);              // ReLU
  gemm_bf16_kernel<true, false, true, false><<<gD, blk256, 0, stream>>>(
      H1, W2T, b2, X1f, out, nullptr, M, D_, FF_);
  // --- final LN (in place on d_out) ---
  layernorm_kernel<<<dim3(M), blk256, 0, stream>>>(out, g1, be1, out, nullptr);
}
